// CausalSelfAttention_12549894439449
// MI455X (gfx1250) — matrix operand order repaired and checked by reading, not compiled, not run on hardware
//
#include <hip/hip_runtime.h>
#include <hip/hip_bf16.h>

typedef __attribute__((ext_vector_type(16))) _Float16 v16h;
typedef __attribute__((ext_vector_type(8)))  float    v8f;
typedef __attribute__((ext_vector_type(4)))  int      v4i;

constexpr int kB = 4, kT = 2048, kC = 2048, kH = 16, kD = 128;

// ---------------------------------------------------------------------------
// gfx1250 async global->LDS copy (ASYNCcnt path), with sync fallback
// ---------------------------------------------------------------------------
#if defined(__has_builtin)
#if __has_builtin(__builtin_amdgcn_global_load_async_to_lds_b128)
#define USE_ASYNC_LDS 1
#endif
#endif
#ifndef USE_ASYNC_LDS
#define USE_ASYNC_LDS 0
#endif

#if USE_ASYNC_LDS
typedef __attribute__((address_space(1))) v4i* gptr16_t;  // global int4*
typedef __attribute__((address_space(3))) v4i* lptr16_t;  // LDS int4*
__device__ __forceinline__ void async_copy16(const void* g, void* l) {
  __builtin_amdgcn_global_load_async_to_lds_b128((gptr16_t)g, (lptr16_t)l, 0, 0);
}
__device__ __forceinline__ void wait_async(int pending) {
#if __has_builtin(__builtin_amdgcn_s_wait_asynccnt)
  if (pending) __builtin_amdgcn_s_wait_asynccnt(16);
  else         __builtin_amdgcn_s_wait_asynccnt(0);
#else
  if (pending) asm volatile("s_wait_asynccnt 0x10" ::: "memory");
  else         asm volatile("s_wait_asynccnt 0x0" ::: "memory");
#endif
}
#endif

// ---------------------------------------------------------------------------
// WMMA helpers (CDNA5: V_WMMA_F32_16X16X32_F16, D = A(16x32) * B(32x16) + C)
// ---------------------------------------------------------------------------
__device__ __forceinline__ v8f wmma_f16(v16h a, v16h b, v8f c) {
  return __builtin_amdgcn_wmma_f32_16x16x32_f16(
      /*neg_a=*/false, a, /*neg_b=*/false, b,
      /*c_mod=*/(short)0, c, /*reuse_a=*/false, /*reuse_b=*/false);
}

// A fragment: 16x32 f16, row-major source with `stride` halves per row.
__device__ __forceinline__ v16h load_frag_a(const _Float16* p0, int row_base,
                                            int stride) {
  const int lane = threadIdx.x & 31;
  const _Float16* p = p0 + (row_base + (lane & 15)) * stride;
  const int kb = (lane >> 4) * 8;
  union { unsigned u[8]; v16h v; } t;
#pragma unroll
  for (int i = 0; i < 8; ++i) {
    const int kk = ((i < 4) ? (2 * i) : (16 + 2 * (i - 4))) + kb;
    t.u[i] = *(const unsigned*)(p + kk);
  }
  return t.v;
}

// B fragment: 32x16 f16 from a K-major (transposed, [N][K]) tile.
__device__ __forceinline__ v16h load_frag_b(const _Float16* p0, int col_base,
                                            int stride) {
  const int lane = threadIdx.x & 31;
  const _Float16* p = p0 + (col_base + (lane & 15)) * stride + (lane >> 4) * 8;
  union { uint4 u[2]; v16h v; } t;
  t.u[0] = *(const uint4*)(p);
  t.u[1] = *(const uint4*)(p + 16);
  return t.v;
}

// ---------------------------------------------------------------------------
// Kernel 1: QKV projection.  qkv = x @ w_qkv^T + b_qkv
//   Double-buffered LDS; next tile's global loads issued before the WMMAs.
//   Writes Q,K f16 [B,H,T,D], V f16 [B,H,D,T] (pre-transposed for P@V).
// ---------------------------------------------------------------------------
__global__ __launch_bounds__(256) void qkv_kernel(
    const float* __restrict__ x, const float* __restrict__ w,
    const float* __restrict__ bias, _Float16* __restrict__ qbuf,
    _Float16* __restrict__ kbuf, _Float16* __restrict__ vbuf) {
  constexpr int BM = 128, BN = 256, BK = 32, AS = BK + 8, BS = BK + 8;
  __shared__ _Float16 As[2][BM * AS];
  __shared__ _Float16 Bs[2][BN * BS];
  const int tid = threadIdx.x;
  const int m0 = blockIdx.y * BM;
  const int n0 = blockIdx.x * BN;
  const int wid = tid >> 5;
  const int wy = wid & 1, wx = wid >> 1;  // 2x4 wave grid, 64x64 per wave
  const int arow = tid >> 3, acol = (tid & 7) * 4;

  v8f acc[4][4] = {};
  float4 ra[4], rb[8];

  auto gload = [&](int k0) {
#pragma unroll
    for (int r = 0; r < 4; ++r)
      ra[r] = *(const float4*)(x + (size_t)(m0 + arow + r * 32) * kC + k0 + acol);
#pragma unroll
    for (int r = 0; r < 8; ++r)
      rb[r] = *(const float4*)(w + (size_t)(n0 + arow + r * 32) * kC + k0 + acol);
  };
  auto sstore = [&](int buf) {
#pragma unroll
    for (int r = 0; r < 4; ++r) {
      _Float16* dst = As[buf] + (arow + r * 32) * AS + acol;
      dst[0] = (_Float16)ra[r].x; dst[1] = (_Float16)ra[r].y;
      dst[2] = (_Float16)ra[r].z; dst[3] = (_Float16)ra[r].w;
    }
#pragma unroll
    for (int r = 0; r < 8; ++r) {
      _Float16* dst = Bs[buf] + (arow + r * 32) * BS + acol;
      dst[0] = (_Float16)rb[r].x; dst[1] = (_Float16)rb[r].y;
      dst[2] = (_Float16)rb[r].z; dst[3] = (_Float16)rb[r].w;
    }
  };

  gload(0);
  sstore(0);
  for (int k0 = 0, it = 0; k0 < kC; k0 += BK, ++it) {
    const int cur = it & 1;
    __syncthreads();
    const bool nxt = (k0 + BK) < kC;
    if (nxt) gload(k0 + BK);  // overlap global latency with WMMAs below
    v16h af[4], bf[4];
#pragma unroll
    for (int i = 0; i < 4; ++i) af[i] = load_frag_a(As[cur], wy * 64 + i * 16, AS);
#pragma unroll
    for (int j = 0; j < 4; ++j) bf[j] = load_frag_b(Bs[cur], wx * 64 + j * 16, BS);
#pragma unroll
    for (int i = 0; i < 4; ++i)
#pragma unroll
      for (int j = 0; j < 4; ++j) acc[i][j] = wmma_f16(af[i], bf[j], acc[i][j]);
    if (nxt) sstore(cur ^ 1);
  }

  const int lane = tid & 31;
  const int colL = lane & 15, rowH = (lane >> 4) * 8;
#pragma unroll
  for (int i = 0; i < 4; ++i) {
#pragma unroll
    for (int j = 0; j < 4; ++j) {
      const int n = n0 + wx * 64 + j * 16 + colL;
      const float bv = bias[n];
      const int type = n >> 11;  // 0=Q 1=K 2=V
      const int c = n & (kC - 1);
      const int hh = c >> 7, dd = c & (kD - 1);
#pragma unroll
      for (int r = 0; r < 8; ++r) {
        const int m = m0 + wy * 64 + i * 16 + r + rowH;
        const int bb = m >> 11, tt = m & (kT - 1);
        const float v = acc[i][j][r] + bv;
        if (type == 0)
          qbuf[(((size_t)bb * kH + hh) * kT + tt) * kD + dd] = (_Float16)v;
        else if (type == 1)
          kbuf[(((size_t)bb * kH + hh) * kT + tt) * kD + dd] = (_Float16)v;
        else
          vbuf[(((size_t)bb * kH + hh) * kD + dd) * kT + tt] = (_Float16)v;
      }
    }
  }
}

// ---------------------------------------------------------------------------
// Kernel 2: flash attention.  4 waves/block, each wave owns 16 query rows.
//   Double-buffered KV tiles staged with async global->LDS copies.
// ---------------------------------------------------------------------------
__global__ __launch_bounds__(128) void attn_kernel(
    const _Float16* __restrict__ qbuf, const _Float16* __restrict__ kbuf,
    const _Float16* __restrict__ vbuf, _Float16* __restrict__ obuf) {
  constexpr int KSs = kD + 8;  // K tile [64][128] stride (halves)
  constexpr int VSs = 64 + 8;  // V tile [128][64] stride
  constexpr int PSs = 64 + 8;  // P tile [16][64] stride per wave
  __shared__ _Float16 Ks[2][64 * KSs];
  __shared__ _Float16 Vs[2][kD * VSs];
  __shared__ _Float16 Ps[4 * 16 * PSs];

  const int tid = threadIdx.x;
  const int lane = tid & 31, wave = tid >> 5;
  const int bh = blockIdx.y;  // b*H + h
  const int qt = blockIdx.x;
  const int qrow = qt * 64 + wave * 16;
  const float scale = 0.08838834764831845f;  // 1/sqrt(128)

  // Q fragments resident in registers: 16 rows x 128 dims = 4 A-frags
  const _Float16* qbase = qbuf + ((size_t)bh * kT + qrow) * kD;
  v16h aq[4];
#pragma unroll
  for (int kc = 0; kc < 4; ++kc) aq[kc] = load_frag_a(qbase + kc * 32, 0, kD);

  v8f oacc[8] = {};
  float mrow[8], lrow[8];
#pragma unroll
  for (int r = 0; r < 8; ++r) { mrow[r] = -3.0e38f; lrow[r] = 0.0f; }

  const int colL = lane & 15, rowH = (lane >> 4) * 8;
  _Float16* Pw = Ps + wave * 16 * PSs;

  auto stage = [&](int kt, int buf) {  // 16 x 16B per thread
    const _Float16* ksrc = kbuf + ((size_t)bh * kT + kt * 64) * kD;
    const _Float16* vsrc = vbuf + (size_t)bh * kD * kT + kt * 64;
#pragma unroll
    for (int p = 0; p < 8; ++p) {  // K tile: 64x128 halves
      const int idx = tid + p * 128;
      const int row = idx >> 4, c = (idx & 15) * 8;
#if USE_ASYNC_LDS
      async_copy16(ksrc + row * kD + c, Ks[buf] + row * KSs + c);
#else
      *(uint4*)(Ks[buf] + row * KSs + c) = *(const uint4*)(ksrc + row * kD + c);
#endif
    }
#pragma unroll
    for (int p = 0; p < 8; ++p) {  // V tile: 128x64 halves (d-major)
      const int idx = tid + p * 128;
      const int row = idx >> 3, c = (idx & 7) * 8;
#if USE_ASYNC_LDS
      async_copy16(vsrc + (size_t)row * kT + c, Vs[buf] + row * VSs + c);
#else
      *(uint4*)(Vs[buf] + row * VSs + c) = *(const uint4*)(vsrc + (size_t)row * kT + c);
#endif
    }
  };

  stage(0, 0);
  for (int kt = 0; kt <= qt; ++kt) {
    const int cur = kt & 1;
    if (kt < qt) stage(kt + 1, cur ^ 1);  // prefetch next tile into other buffer
#if USE_ASYNC_LDS
    wait_async(kt < qt ? 16 : 0);  // retire current tile (FIFO order)
#endif
    __syncthreads();

    // S = Q @ K^T  (16 x 64), 16 WMMAs
    v8f sacc[4] = {};
#pragma unroll
    for (int j = 0; j < 4; ++j)
#pragma unroll
      for (int kc = 0; kc < 4; ++kc)
        sacc[j] =
            wmma_f16(aq[kc], load_frag_b(Ks[cur] + kc * 32, j * 16, KSs), sacc[j]);

    // scale + causal mask + running row max
    float pv[4][8], rmax[8];
#pragma unroll
    for (int r = 0; r < 8; ++r) rmax[r] = -3.0e38f;
#pragma unroll
    for (int j = 0; j < 4; ++j) {
      const int kv = kt * 64 + j * 16 + colL;
#pragma unroll
      for (int r = 0; r < 8; ++r) {
        const int qr = qrow + r + rowH;
        float v = sacc[j][r] * scale;
        v = (kv > qr) ? -1.0e30f : v;
        pv[j][r] = v;
        rmax[r] = fmaxf(rmax[r], v);
      }
    }
    // online softmax update (16-lane reductions; halves hold disjoint rows)
#pragma unroll
    for (int r = 0; r < 8; ++r) {
#pragma unroll
      for (int s = 1; s < 16; s <<= 1)
        rmax[r] = fmaxf(rmax[r], __shfl_xor(rmax[r], s, 32));
      const float mnew = fmaxf(mrow[r], rmax[r]);
      const float alpha = __expf(mrow[r] - mnew);
      mrow[r] = mnew;
      lrow[r] *= alpha;
#pragma unroll
      for (int d = 0; d < 8; ++d) oacc[d][r] *= alpha;
    }
    // P = exp(S - m); stage to LDS in A-matrix layout
#pragma unroll
    for (int j = 0; j < 4; ++j) {
#pragma unroll
      for (int r = 0; r < 8; ++r) {
        const float e = __expf(pv[j][r] - mrow[r]);
        pv[j][r] = e;
        Pw[(r + rowH) * PSs + j * 16 + colL] = (_Float16)e;
      }
    }
#pragma unroll
    for (int r = 0; r < 8; ++r) {
      float s = (pv[0][r] + pv[1][r]) + (pv[2][r] + pv[3][r]);
#pragma unroll
      for (int sh = 1; sh < 16; sh <<= 1) s += __shfl_xor(s, sh, 32);
      lrow[r] += s;
    }
    __syncthreads();

    // O += P @ V  (16 x 128), 16 WMMAs
#pragma unroll
    for (int d = 0; d < 8; ++d)
#pragma unroll
      for (int pc = 0; pc < 2; ++pc)
        oacc[d] = wmma_f16(load_frag_a(Pw + pc * 32, 0, PSs),
                           load_frag_b(Vs[cur] + pc * 32, d * 16, VSs), oacc[d]);
    __syncthreads();  // all waves done with buffers before restaging
  }

  // normalize and emit O as f16 [B,T,C]
  const int b = bh >> 4, h = bh & 15;
#pragma unroll
  for (int r = 0; r < 8; ++r) {
    const float inv = 1.0f / lrow[r];
    const size_t m = (size_t)b * kT + qrow + r + rowH;
#pragma unroll
    for (int d = 0; d < 8; ++d)
      obuf[m * kC + h * kD + d * 16 + colL] = (_Float16)(oacc[d][r] * inv);
  }
}

// ---------------------------------------------------------------------------
// Kernel 3: output projection.  out = O @ w_proj^T + b_proj (fp32 out)
// ---------------------------------------------------------------------------
__global__ __launch_bounds__(256) void proj_kernel(
    const _Float16* __restrict__ a, const float* __restrict__ w,
    const float* __restrict__ bias, float* __restrict__ out) {
  constexpr int BM = 128, BN = 256, BK = 32, AS = BK + 8, BS = BK + 8;
  __shared__ _Float16 As[2][BM * AS];
  __shared__ _Float16 Bs[2][BN * BS];
  const int tid = threadIdx.x;
  const int m0 = blockIdx.y * BM;
  const int n0 = blockIdx.x * BN;
  const int wid = tid >> 5;
  const int wy = wid & 1, wx = wid >> 1;
  const int arow = tid >> 3, acol = (tid & 7) * 4;
  v8f acc[4][4] = {};
  uint4 raH[2];
  float4 rb[8];

  auto gload = [&](int k0) {
#pragma unroll
    for (int p = 0; p < 2; ++p) {  // A tile 128x32 halves (already f16)
      const int idx = tid + p * 256;
      const int row = idx >> 2, c = (idx & 3) * 8;
      raH[p] = *(const uint4*)(a + (size_t)(m0 + row) * kC + k0 + c);
    }
#pragma unroll
    for (int r = 0; r < 8; ++r)
      rb[r] = *(const float4*)(w + (size_t)(n0 + arow + r * 32) * kC + k0 + acol);
  };
  auto sstore = [&](int buf) {
#pragma unroll
    for (int p = 0; p < 2; ++p) {
      const int idx = tid + p * 256;
      const int row = idx >> 2, c = (idx & 3) * 8;
      *(uint4*)(As[buf] + row * AS + c) = raH[p];
    }
#pragma unroll
    for (int r = 0; r < 8; ++r) {
      _Float16* dst = Bs[buf] + (arow + r * 32) * BS + acol;
      dst[0] = (_Float16)rb[r].x; dst[1] = (_Float16)rb[r].y;
      dst[2] = (_Float16)rb[r].z; dst[3] = (_Float16)rb[r].w;
    }
  };

  gload(0);
  sstore(0);
  for (int k0 = 0, it = 0; k0 < kC; k0 += BK, ++it) {
    const int cur = it & 1;
    __syncthreads();
    const bool nxt = (k0 + BK) < kC;
    if (nxt) gload(k0 + BK);
    v16h af[4], bf[4];
#pragma unroll
    for (int i = 0; i < 4; ++i) af[i] = load_frag_a(As[cur], wy * 64 + i * 16, AS);
#pragma unroll
    for (int j = 0; j < 4; ++j) bf[j] = load_frag_b(Bs[cur], wx * 64 + j * 16, BS);
#pragma unroll
    for (int i = 0; i < 4; ++i)
#pragma unroll
      for (int j = 0; j < 4; ++j) acc[i][j] = wmma_f16(af[i], bf[j], acc[i][j]);
    if (nxt) sstore(cur ^ 1);
  }

  const int lane = tid & 31;
  const int colL = lane & 15, rowH = (lane >> 4) * 8;
#pragma unroll
  for (int i = 0; i < 4; ++i)
#pragma unroll
    for (int j = 0; j < 4; ++j) {
      const int n = n0 + wx * 64 + j * 16 + colL;
      const float bv = bias[n];
#pragma unroll
      for (int r = 0; r < 8; ++r) {
        const int m = m0 + wy * 64 + i * 16 + r + rowH;
        out[(size_t)m * kC + n] = acc[i][j][r] + bv;
      }
    }
}

// ---------------------------------------------------------------------------
extern "C" void kernel_launch(void* const* d_in, const int* in_sizes, int n_in,
                              void* d_out, int out_size, void* d_ws,
                              size_t ws_size, hipStream_t stream) {
  (void)in_sizes; (void)n_in; (void)out_size; (void)ws_size;
  const float* x      = (const float*)d_in[0];
  const float* w_qkv  = (const float*)d_in[1];
  const float* b_qkv  = (const float*)d_in[2];
  const float* w_proj = (const float*)d_in[3];
  const float* b_proj = (const float*)d_in[4];
  float* out = (float*)d_out;

  const size_t SZ = (size_t)kB * kH * kT * kD;  // 16.78M elements
  _Float16* qbuf = (_Float16*)d_ws;
  _Float16* kbuf = qbuf + SZ;
  _Float16* vbuf = kbuf + SZ;
  _Float16* obuf = vbuf + SZ;

  qkv_kernel<<<dim3(6144 / 256, 8192 / 128), 256, 0, stream>>>(
      x, w_qkv, b_qkv, qbuf, kbuf, vbuf);
  attn_kernel<<<dim3(kT / 64, kB * kH), 128, 0, stream>>>(qbuf, kbuf, vbuf, obuf);
  proj_kernel<<<dim3(kC / 256, 8192 / 128), 256, 0, stream>>>(
      obuf, w_proj, b_proj, out);
}
